// FaceIDAttnProcessor_59785944760508
// MI455X (gfx1250) — matrix operand order repaired and checked by reading, not compiled, not run on hardware
//
#include <hip/hip_runtime.h>
#include <hip/hip_bf16.h>

// Dims (compile-time; match reference setup_inputs)
#define NHEADS 20
#define B_  8
#define S_  4096
#define H_  1280
#define L_  77
#define E_  512
#define C_  768
#define D_  64
#define MP_ 624   // B_*L_ = 616 padded to 39 full 16-row tiles

typedef __bf16          bf16x16 __attribute__((ext_vector_type(16)));
typedef float           f32x8   __attribute__((ext_vector_type(8)));
typedef float           f32x4   __attribute__((ext_vector_type(4)));
typedef unsigned short  u16x16  __attribute__((ext_vector_type(16)));
typedef unsigned short  u16x8   __attribute__((ext_vector_type(8)));
typedef unsigned short  u16x4   __attribute__((ext_vector_type(4)));

// Async global->LDS via inline asm (the clang builtin's params are LangAS
// cuda_device/cuda_shared int4*, unspellable in HIP source).
#if defined(__gfx1250__)
#  define USE_ASYNC 1
#else
#  define USE_ASYNC 0
#endif

__device__ __forceinline__ unsigned short f2bf(float f) {
    return __builtin_bit_cast(unsigned short, (__bf16)f);   // native RNE convert
}

__device__ __forceinline__ f32x8 wmma_bf16(u16x16 a, u16x16 b, f32x8 c) {
    return __builtin_amdgcn_wmma_f32_16x16x32_bf16(
        false, __builtin_bit_cast(bf16x16, a),
        false, __builtin_bit_cast(bf16x16, b),
        (short)0, c, false, false);
}

// Concatenate two 16-B halves into one fragment register block
__device__ __forceinline__ u16x16 cat16(u16x8 lo, u16x8 hi) {
    return __builtin_shufflevector(lo, hi, 0,1,2,3,4,5,6,7,8,9,10,11,12,13,14,15);
}
// Fragment from two contiguous-16B chunks at p and p+off (ushort units)
__device__ __forceinline__ u16x16 frag2(const unsigned short* p, int off) {
    return cat16(*(const u16x8*)p, *(const u16x8*)(p + off));
}
// Convert 8 consecutive f32 -> 8 bf16
__device__ __forceinline__ u16x8 cvt8(const float* p) {
    f32x4 v0 = *(const f32x4*)p;
    f32x4 v1 = *(const f32x4*)(p + 4);
    u16x8 r;
#pragma unroll
    for (int i = 0; i < 4; ++i) { r[i] = f2bf(v0[i]); r[i + 4] = f2bf(v1[i]); }
    return r;
}

// Copy one 16-byte chunk global -> LDS (ASYNCcnt-tracked)
__device__ __forceinline__ void cp16_async(const unsigned short* g, unsigned short* l) {
#if USE_ASYNC
    unsigned lds_addr = (unsigned)(size_t)l;     // low 32 bits of generic = LDS offset
    unsigned long long gaddr = (unsigned long long)(size_t)g;
    asm volatile("global_load_async_to_lds_b128 %0, %1, off"
                 :: "v"(lds_addr), "v"(gaddr)
                 : "memory");
#else
    *(u16x8*)l = *(const u16x8*)g;
#endif
}

template<int N> __device__ __forceinline__ void async_waitle() {
#if USE_ASYNC
    asm volatile("s_wait_asynccnt %0" :: "i"(N) : "memory");
#endif
}

// ---------------------------------------------------------------------------
// Elementwise f32 -> bf16 (4 per thread, vectorized)
// ---------------------------------------------------------------------------
__global__ __launch_bounds__(256)
void cvt_kernel(const float* __restrict__ in, unsigned short* __restrict__ out, int n4) {
    int i = blockIdx.x * 256 + threadIdx.x;
    if (i < n4) {
        f32x4 v = *(const f32x4*)&in[(size_t)i * 4];
        u16x4 o;
        o[0] = f2bf(v[0]); o[1] = f2bf(v[1]); o[2] = f2bf(v[2]); o[3] = f2bf(v[3]);
        *(u16x4*)&out[(size_t)i * 4] = o;
    }
}

// ---------------------------------------------------------------------------
// Weight transpose + convert: W[K][N] f32 -> Wt[N][K] bf16 (32x32 LDS tiles)
// ---------------------------------------------------------------------------
__global__ __launch_bounds__(256)
void wt_kernel(const float* __restrict__ W, unsigned short* __restrict__ Wt, int K, int N) {
    __shared__ unsigned short t[32][34];
    const int k0 = blockIdx.y * 32, n0 = blockIdx.x * 32;
    const int tx = threadIdx.x & 31, ty = threadIdx.x >> 5;   // ty 0..7
#pragma unroll
    for (int i = 0; i < 4; ++i) {
        int k = k0 + ty + i * 8, n = n0 + tx;
        t[ty + i * 8][tx] = (k < K && n < N) ? f2bf(W[(size_t)k * N + n]) : (unsigned short)0;
    }
    __syncthreads();
#pragma unroll
    for (int i = 0; i < 4; ++i) {
        int n = n0 + ty + i * 8, k = k0 + tx;
        if (n < N && k < K) Wt[(size_t)n * K + k] = t[tx][ty + i * 8];
    }
}

// ---------------------------------------------------------------------------
// enc = encoder_hidden_states @ Wfp + bfp  -> [624(pad) x 768] bf16, K=512
// One wave per 16x16 tile; A rows clamped to the 616 real input rows; all
// stores unconditional into the padded workspace.
// ---------------------------------------------------------------------------
__global__ __launch_bounds__(32)
void encproj_kernel(const float* __restrict__ enc_in, const unsigned short* __restrict__ Wfpt,
                    const float* __restrict__ bfp, unsigned short* __restrict__ enc_out) {
    const int lane  = threadIdx.x;
    const int n0    = blockIdx.x * 16;
    const int m0    = blockIdx.y * 16;
    const int M     = B_ * L_;
    const int nl    = lane & 15;
    const int khalf = lane >> 4;
    const int mlane = m0 + nl;
    const int msafe = mlane < M ? mlane : (M - 1);
    const unsigned short* brow = Wfpt + (size_t)(n0 + nl) * E_;
    f32x8 acc = {0.f,0.f,0.f,0.f,0.f,0.f,0.f,0.f};
    for (int kk = 0; kk < E_; kk += 32) {
        const float* ap = enc_in + (size_t)msafe * E_ + kk + (khalf << 3);
        u16x16 a = cat16(cvt8(ap), cvt8(ap + 16));
        u16x16 b = frag2(&brow[kk + (khalf << 3)], 16);
        acc = wmma_bf16(a, b, acc);
    }
    const float bias = bfp[n0 + nl];
#pragma unroll
    for (int r = 0; r < 8; ++r) {
        int m = m0 + r + (khalf << 3);
        enc_out[(size_t)m * C_ + n0 + nl] = f2bf(acc[r] + bias);
    }
}

// ---------------------------------------------------------------------------
// k = enc @ Wk ; v = enc @ Wv  -> [624(pad) x 1280] bf16, K=768
// ---------------------------------------------------------------------------
__global__ __launch_bounds__(32)
void kvproj_kernel(const unsigned short* __restrict__ enc,
                   const unsigned short* __restrict__ Wkt, const unsigned short* __restrict__ Wvt,
                   unsigned short* __restrict__ kout, unsigned short* __restrict__ vout) {
    const int lane  = threadIdx.x;
    const int n0    = blockIdx.x * 16;
    const int m0    = blockIdx.y * 16;
    const unsigned short* Wt  = blockIdx.z ? Wvt : Wkt;
    unsigned short*       out = blockIdx.z ? vout : kout;
    const int nl    = lane & 15;
    const int khalf = lane >> 4;
    const unsigned short* arow = enc + (size_t)(m0 + nl) * C_;     // padded ws, no clamp
    const unsigned short* brow = Wt + (size_t)(n0 + nl) * C_;
    f32x8 acc = {0.f,0.f,0.f,0.f,0.f,0.f,0.f,0.f};
    for (int kk = 0; kk < C_; kk += 32) {
        u16x16 a = frag2(&arow[kk + (khalf << 3)], 16);
        u16x16 b = frag2(&brow[kk + (khalf << 3)], 16);
        acc = wmma_bf16(a, b, acc);
    }
#pragma unroll
    for (int r = 0; r < 8; ++r) {
        int m = m0 + r + (khalf << 3);
        out[(size_t)m * H_ + n0 + nl] = f2bf(acc[r]);
    }
}

// ---------------------------------------------------------------------------
// Big GEMM: C[M,N] = A[M,K] @ Bt[N,K]^T (+bias, +residual)
// A, Bt bf16. 256 threads = 8 waves; block tile 128x128, BK=32.
// Double-buffered async global->LDS staging (4 x b128 chunks per thread/tile).
// LDS row stride 56 ushorts: 16B rows, conflict-free 16-row fragment reads.
// ---------------------------------------------------------------------------
#define GST 56

template<bool RESID>
__global__ __launch_bounds__(256)
void gemm128x128(const unsigned short* __restrict__ A, const unsigned short* __restrict__ Bt,
                 const float* __restrict__ bias, const float* __restrict__ resid,
                 void* __restrict__ Cptr, int M, int N, int K) {
    __shared__ unsigned short As[2][128][GST];
    __shared__ unsigned short Bs[2][128][GST];

    const int tid   = threadIdx.x;
    const int lane  = tid & 31;
    const int w     = tid >> 5;
    const int m0    = blockIdx.y * 128;
    const int n0    = blockIdx.x * 128;
    const int nl    = lane & 15;
    const int khalf = lane >> 4;

    auto stage = [&](int buf, int kk) {
#pragma unroll
        for (int i = 0; i < 2; ++i) {
            int id = i * 256 + tid;                 // 0..511
            int row = id >> 2, c = (id & 3) << 3;   // 4 chunks of 8 ushorts per row
            cp16_async(&A[(size_t)(m0 + row) * K + kk + c], &As[buf][row][c]);
        }
#pragma unroll
        for (int i = 0; i < 2; ++i) {
            int id = i * 256 + tid;
            int row = id >> 2, c = (id & 3) << 3;
            cp16_async(&Bt[(size_t)(n0 + row) * K + kk + c], &Bs[buf][row][c]);
        }
    };

    f32x8 acc[8];
#pragma unroll
    for (int i = 0; i < 8; ++i) acc[i] = (f32x8){0.f,0.f,0.f,0.f,0.f,0.f,0.f,0.f};

    const int nsteps = K / 32;
    stage(0, 0);
    for (int s = 0; s < nsteps; ++s) {
        const int buf = s & 1;
        if (s + 1 < nsteps) {
            stage(buf ^ 1, (s + 1) * 32);   // prefetch next tile into other buffer
            async_waitle<4>();              // drain current tile, keep next 4 in flight
        } else {
            async_waitle<0>();
        }
        __syncthreads();

        u16x16 a = frag2(&As[buf][(w << 4) + nl][khalf << 3], 16);
#pragma unroll
        for (int nt = 0; nt < 8; ++nt) {
            u16x16 b = frag2(&Bs[buf][(nt << 4) + nl][khalf << 3], 16);
            acc[nt] = wmma_bf16(a, b, acc[nt]);
        }
        __syncthreads();
    }

#pragma unroll
    for (int nt = 0; nt < 8; ++nt) {
        int n = n0 + (nt << 4) + nl;
        float bv = bias[n];
#pragma unroll
        for (int r = 0; r < 8; ++r) {
            int m = m0 + (w << 4) + r + (khalf << 3);
            float val = acc[nt][r] + bv;
            if (RESID) {
                val += resid[(size_t)m * N + n];
                ((float*)Cptr)[(size_t)m * N + n] = val;
            } else {
                ((unsigned short*)Cptr)[(size_t)m * N + n] = f2bf(val);
            }
        }
    }
}

// ---------------------------------------------------------------------------
// Attention: per (b, h, 64-row chunk of S). 128 threads = 4 waves; each wave
// owns a 16-row strip. L=77 padded to 80 (scores N) / 96 (PV K).
// k tile async-copied to LDS; scores -> masked softmax -> PV, all WMMA bf16.
// ---------------------------------------------------------------------------
__global__ __launch_bounds__(128)
void attn_kernel(const unsigned short* __restrict__ q,
                 const unsigned short* __restrict__ k,
                 const unsigned short* __restrict__ v,
                 unsigned short* __restrict__ attn) {
    __shared__ unsigned short ks[80][72];     // [l][d], stride 72 (16B mult, conflict-free)
    __shared__ unsigned short vsT[64][104];   // [d][l] transposed
    __shared__ unsigned short ps[4][16][104]; // per-wave P (A matrix for PV)

    const int tid   = threadIdx.x;
    const int lane  = tid & 31;
    const int w     = tid >> 5;
    const int b     = blockIdx.z;
    const int h     = blockIdx.y;
    const int s0    = blockIdx.x * 64;
    const int nl    = lane & 15;
    const int khalf = lane >> 4;
    const float scale = 0.125f;               // 1/sqrt(64)

    // zero pad rows of ks (l = 77..79)
    for (int idx = tid; idx < 3 * 64; idx += 128)
        ks[77 + (idx >> 6)][idx & 63] = 0;
    // async copy k rows 0..76 (8 x 16B chunks per row)
    for (int idx = tid; idx < 77 * 8; idx += 128) {
        int l = idx >> 3, c = (idx & 7) << 3;
        cp16_async(&k[((size_t)b * L_ + l) * H_ + h * D_ + c], &ks[l][c]);
    }
    // v transposed into LDS (manual; transpose can't be a straight byte copy)
    for (int idx = tid; idx < 80 * 64; idx += 128) {
        int l = idx >> 6, d = idx & 63;
        vsT[d][l] = (l < L_) ? v[((size_t)b * L_ + l) * H_ + h * D_ + d] : (unsigned short)0;
    }
    for (int idx = tid; idx < 64 * 16; idx += 128)   // zero PV K-pad cols 80..95
        vsT[idx >> 4][80 + (idx & 15)] = 0;
    for (int idx = tid; idx < 4 * 16 * 16; idx += 128)  // zero P pad cols 80..95
        ps[idx >> 8][(idx >> 4) & 15][80 + (idx & 15)] = 0;
    async_waitle<0>();
    __syncthreads();

    // q A-fragments for K-tiles d=0..31 and d=32..63 (q already bf16)
    const unsigned short* qp = q + ((size_t)b * S_ + (s0 + (w << 4) + nl)) * H_ + h * D_
                                 + (khalf << 3);
    u16x16 a0 = frag2(qp, 16);
    u16x16 a1 = frag2(qp + 32, 16);

    // scores: 5 N-tiles over L (padded 80)
    float sc[5][8];
#pragma unroll
    for (int t = 0; t < 5; ++t) {
        int l = (t << 4) + nl;
        u16x16 b0 = frag2(&ks[l][khalf << 3], 16);
        u16x16 b1 = frag2(&ks[l][32 + (khalf << 3)], 16);
        f32x8 c = {0.f,0.f,0.f,0.f,0.f,0.f,0.f,0.f};
        c = wmma_bf16(a0, b0, c);
        c = wmma_bf16(a1, b1, c);
        bool valid = ((t << 4) + nl) < L_;
#pragma unroll
        for (int r = 0; r < 8; ++r) sc[t][r] = valid ? c[r] * scale : -1e30f;
    }

    // row softmax: C-frag row = r + khalf*8; its 16 cols live in the 16 lanes
    // of this half-wave -> shfl_xor reductions with width 16.
    float rowinv[8];
#pragma unroll
    for (int r = 0; r < 8; ++r) {
        float m = -1e30f;
#pragma unroll
        for (int t = 0; t < 5; ++t) m = fmaxf(m, sc[t][r]);
#pragma unroll
        for (int off = 8; off >= 1; off >>= 1) m = fmaxf(m, __shfl_xor(m, off, 16));
        float s = 0.f;
#pragma unroll
        for (int t = 0; t < 5; ++t) { float p = __expf(sc[t][r] - m); sc[t][r] = p; s += p; }
#pragma unroll
        for (int off = 8; off >= 1; off >>= 1) s += __shfl_xor(s, off, 16);
        rowinv[r] = 1.f / s;
    }

    // write P (bf16) to LDS for re-read in A-fragment layout
#pragma unroll
    for (int t = 0; t < 5; ++t)
#pragma unroll
        for (int r = 0; r < 8; ++r)
            ps[w][r + (khalf << 3)][(t << 4) + nl] = f2bf(sc[t][r]);
    __syncthreads();

    // out = P @ v : K = 96 (3 tiles of 32), N = 64 (4 tiles of 16)
    f32x8 oacc[4];
#pragma unroll
    for (int i = 0; i < 4; ++i) oacc[i] = (f32x8){0.f,0.f,0.f,0.f,0.f,0.f,0.f,0.f};
#pragma unroll
    for (int kt = 0; kt < 3; ++kt) {
        u16x16 a = frag2(&ps[w][nl][(kt << 5) + (khalf << 3)], 16);
#pragma unroll
        for (int nt = 0; nt < 4; ++nt) {
            u16x16 bfr = frag2(&vsT[(nt << 4) + nl][(kt << 5) + (khalf << 3)], 16);
            oacc[nt] = wmma_bf16(a, bfr, oacc[nt]);
        }
    }

    // normalize + store bf16
#pragma unroll
    for (int nt = 0; nt < 4; ++nt)
#pragma unroll
        for (int r = 0; r < 8; ++r) {
            int m = s0 + (w << 4) + r + (khalf << 3);
            int d = (nt << 4) + nl;
            attn[((size_t)b * S_ + m) * H_ + h * D_ + d] = f2bf(oacc[nt][r] * rowinv[r]);
        }
}

// ---------------------------------------------------------------------------
extern "C" void kernel_launch(void* const* d_in, const int* in_sizes, int n_in,
                              void* d_out, int out_size, void* d_ws, size_t ws_size,
                              hipStream_t stream) {
    (void)in_sizes; (void)n_in; (void)out_size; (void)ws_size;
    const float* hs   = (const float*)d_in[0];
    const float* ench = (const float*)d_in[1];
    const float* Wq   = (const float*)d_in[2];
    const float* bq   = (const float*)d_in[3];
    const float* Wfp  = (const float*)d_in[4];
    const float* bfp  = (const float*)d_in[5];
    const float* Wk   = (const float*)d_in[6];
    const float* Wv   = (const float*)d_in[7];
    const float* Wout = (const float*)d_in[8];
    const float* bout = (const float*)d_in[9];
    float* out = (float*)d_out;

    char* ws = (char*)d_ws;
    size_t off = 0;
    auto salloc = [&](size_t elems) {
        char* p = ws + off;
        off += (elems * 2 + 255) & ~(size_t)255;
        return (unsigned short*)p;
    };
    unsigned short* hs_bf   = salloc((size_t)B_ * S_ * H_);
    unsigned short* q_bf    = salloc((size_t)B_ * S_ * H_);
    unsigned short* attn_bf = salloc((size_t)B_ * S_ * H_);
    unsigned short* enc_bf  = salloc((size_t)MP_ * C_);     // padded rows
    unsigned short* k_bf    = salloc((size_t)MP_ * H_);
    unsigned short* v_bf    = salloc((size_t)MP_ * H_);
    unsigned short* Wqt     = salloc((size_t)H_ * H_);
    unsigned short* Wfpt    = salloc((size_t)E_ * C_);
    unsigned short* Wkt     = salloc((size_t)C_ * H_);
    unsigned short* Wvt     = salloc((size_t)C_ * H_);
    unsigned short* Woutt   = salloc((size_t)H_ * H_);

    // --- one-time precision/layout prep ---
    {
        int n4 = (B_ * S_ * H_) / 4;
        cvt_kernel<<<dim3((n4 + 255) / 256), 256, 0, stream>>>(hs, hs_bf, n4);
        wt_kernel<<<dim3(H_ / 32, H_ / 32), 256, 0, stream>>>(Wq,   Wqt,   H_, H_);
        wt_kernel<<<dim3(C_ / 32, E_ / 32), 256, 0, stream>>>(Wfp,  Wfpt,  E_, C_);
        wt_kernel<<<dim3(H_ / 32, C_ / 32), 256, 0, stream>>>(Wk,   Wkt,   C_, H_);
        wt_kernel<<<dim3(H_ / 32, C_ / 32), 256, 0, stream>>>(Wv,   Wvt,   C_, H_);
        wt_kernel<<<dim3(H_ / 32, H_ / 32), 256, 0, stream>>>(Wout, Woutt, H_, H_);
    }

    const int MT = MP_ / 16;   // 39 row tiles

    encproj_kernel<<<dim3(C_ / 16, MT), 32, 0, stream>>>(ench, Wfpt, bfp, enc_bf);
    kvproj_kernel<<<dim3(H_ / 16, MT, 2), 32, 0, stream>>>(enc_bf, Wkt, Wvt, k_bf, v_bf);
    gemm128x128<false><<<dim3(H_ / 128, (B_ * S_) / 128), 256, 0, stream>>>(
        hs_bf, Wqt, bq, nullptr, q_bf, B_ * S_, H_, H_);
    attn_kernel<<<dim3(S_ / 64, NHEADS, B_), 128, 0, stream>>>(q_bf, k_bf, v_bf, attn_bf);
    gemm128x128<true><<<dim3(H_ / 128, (B_ * S_) / 128), 256, 0, stream>>>(
        attn_bf, Woutt, bout, hs, out, B_ * S_, H_, H_);
}
